// Packetbnn_16114717294826
// MI455X (gfx1250) — matrix operand order repaired and checked by reading, not compiled, not run on hardware
//
#include <hip/hip_runtime.h>

// CDNA5 / gfx1250 fused binary-NN:
//   out[b] = sum_n bin(softsign( sum_k bin(x[b,k]) * bin(W[n,k]) )) * bin(lw[n])
//
// Exact arithmetic shortcuts (all branch-free, 2-3 VALU ops each):
//   * y is always a multiple of 0.25  =>  bin(softsign(y)) == med3(2y-1.5, 0, 1)
//   * bin(t) == med3(fma(2^30, t-0.5, 0.5), 0, 1): d=0 -> 0.5; |d|>=2^-25 saturates.
//
// Kernel 0 (one block): binarize conv_w into the WMMA B-fragment lane layout and
//   binarize/pad lin_w into d_ws (L2-resident, 33 KB).
// Kernel 1 (2048 blocks): stream x once (126 MB, TH=NT), binarize to f16 into LDS,
//   GEMM on v_wmma_f32_16x16x32_f16, fused threshold+GEMV epilogue, NT stores.

typedef __attribute__((ext_vector_type(16))) _Float16 v16h;
typedef __attribute__((ext_vector_type(8)))  _Float16 v8h;
typedef __attribute__((ext_vector_type(8)))  float    v8f;
typedef __attribute__((ext_vector_type(4)))  float    f4;

#define K_DIM   120
#define N_DIM   120
#define ROWS_PER_BLOCK 128
#define THREADS 256
#define LDA     136                    // A row stride (halfs): 272B = 68 dwords (17 x 16B) -> conflict-free column gather
#define LB_HALFS (32 * 32 * 16)        // 8 ntiles * 4 ksteps fragments, 32 lanes x 16 halfs (32 KB)
#define LA_HALFS (ROWS_PER_BLOCK * LDA)
#define WS_LIN_OFF (LB_HALFS * 2)

__device__ __forceinline__ float binf(float t) {
    // exact ((sign(t-0.5)+1)/2): branchy reference form (prep kernel only)
    float d = t - 0.5f;
    return d > 0.f ? 1.f : (d < 0.f ? 0.f : 0.5f);
}

__device__ __forceinline__ float binf_fast(float t) {
    // exact, branch-free: sub + fma + med3
    float d = t - 0.5f;                            // sign-exact; |d|>=2^-25 when nonzero
    return __builtin_amdgcn_fmed3f(__builtin_fmaf(1073741824.0f, d, 0.5f), 0.f, 1.f);
}

__device__ __forceinline__ float bin_softsign(float y) {
    // exact for y in 0.25*Z: fma + med3 (softsign monotone, crosses 0.5 at y==1)
    return __builtin_amdgcn_fmed3f(__builtin_fmaf(2.0f, y, -1.5f), 0.f, 1.f);
}

// ---------------- Kernel 0: one-time weight prep into workspace ----------------
__global__ __launch_bounds__(THREADS)
void bnn_prep_kernel(const float* __restrict__ conv_w,
                     const float* __restrict__ lin_w,
                     _Float16* __restrict__ wsB,
                     float* __restrict__ wsL)
{
    const int tid = threadIdx.x;
    // B (32x16, f16): lane half selects K block of 16; elem e -> K = 16*(lane/16)+e, N = lane&15.
    for (int i = tid; i < LB_HALFS; i += THREADS) {
        int e     = i & 15;
        int lane  = (i >> 4) & 31;
        int frag  = i >> 9;                 // ntile*4 + kstep
        int ntile = frag >> 2, kstep = frag & 3;
        int n = ntile * 16 + (lane & 15);
        int k = kstep * 32 + ((lane >> 4) << 4) + e;
        float v = 0.f;
        if (n < N_DIM && k < K_DIM) v = binf(conv_w[n * K_DIM + k]);
        wsB[i] = (_Float16)v;
    }
    if (tid < 128) wsL[tid] = (tid < N_DIM) ? binf(lin_w[tid]) : 0.f;
}

// ---------------- Kernel 1: fused streaming GEMM + epilogue ----------------
__global__ __launch_bounds__(THREADS)
void bnn_fused_kernel(const float* __restrict__ x,
                      const _Float16* __restrict__ wsB,
                      const float* __restrict__ wsL,
                      float* __restrict__ out)
{
    extern __shared__ __align__(32) char smem_raw[];
    _Float16* lB = (_Float16*)smem_raw;                        // 32768 B, 32B-aligned fragments
    _Float16* lA = (_Float16*)(smem_raw + LB_HALFS * 2);       // 34816 B

    const int tid = threadIdx.x;

    // ---- bulk-copy prebuilt B fragments (L2-resident) into LDS ----
    for (int i = tid; i < LB_HALFS / 8; i += THREADS)
        ((f4*)lB)[i] = ((const f4*)wsB)[i];

    // ---- stream x: coalesced non-temporal float4 loads, binarize to f16 into LDS ----
    const float* xblk = x + (size_t)blockIdx.x * (ROWS_PER_BLOCK * K_DIM);
    for (int i = tid * 4; i < ROWS_PER_BLOCK * K_DIM; i += THREADS * 4) {
        f4 v = __builtin_nontemporal_load((const f4*)(xblk + i)); // 120%4==0: no row straddle
        int r = i / K_DIM, c = i - r * K_DIM;
        _Float16* dst = &lA[r * LDA + c];
        dst[0] = (_Float16)binf_fast(v[0]);
        dst[1] = (_Float16)binf_fast(v[1]);
        dst[2] = (_Float16)binf_fast(v[2]);
        dst[3] = (_Float16)binf_fast(v[3]);
    }
    {   // zero the K padding columns 120..127
        int r = tid >> 1, c = K_DIM + (tid & 1) * 4;
        _Float16* dst = &lA[r * LDA + c];
        dst[0] = dst[1] = dst[2] = dst[3] = (_Float16)0.f;
    }
    __syncthreads();

    // ---- per-wave 16-row tile ----
    const int wave = tid >> 5, lane = tid & 31;
    const int halfSel = lane >> 4, mrow = lane & 15;
    const int arow = wave * 16 + mrow;

    // A (16x32, f16): lane = M; elems 0..15 -> K = k0..k0+15,
    // k0 = kstep*32 + half*16.
    v16h afrag[4];
#pragma unroll
    for (int kstep = 0; kstep < 4; ++kstep) {
        int k0 = kstep * 32 + halfSel * 16;
        v8h lo = *(const v8h*)&lA[arow * LDA + k0];
        v8h hi = *(const v8h*)&lA[arow * LDA + k0 + 8];
#pragma unroll
        for (int j = 0; j < 8; ++j) { afrag[kstep][j] = lo[j]; afrag[kstep][j + 8] = hi[j]; }
    }

    float rowsum[8];
#pragma unroll
    for (int e = 0; e < 8; ++e) rowsum[e] = 0.f;

#pragma unroll
    for (int ntile = 0; ntile < 8; ++ntile) {
        v8f acc = {};
#pragma unroll
        for (int kstep = 0; kstep < 4; ++kstep) {
            v16h b = *(const v16h*)&lB[((ntile * 4 + kstep) * 32 + lane) * 16];
            acc = __builtin_amdgcn_wmma_f32_16x16x32_f16(
                false, afrag[kstep], false, b, (short)0, acc, false, false);
        }
        // C/D elem e holds (row = e + 8*half, col = ntile*16 + mrow)
        float lwb = wsL[ntile * 16 + mrow];
#pragma unroll
        for (int e = 0; e < 8; ++e)
            rowsum[e] = __builtin_fmaf(bin_softsign(acc[e]), lwb, rowsum[e]);
    }

    // Reduce the 16 column-lanes of each half-wave
#pragma unroll
    for (int off = 1; off < 16; off <<= 1) {
#pragma unroll
        for (int e = 0; e < 8; ++e)
            rowsum[e] += __shfl_xor(rowsum[e], off, 32);
    }

    if (mrow == 0) {
        int g = blockIdx.x * ROWS_PER_BLOCK + wave * 16 + halfSel * 8;
#pragma unroll
        for (int e = 0; e < 8; ++e)
            __builtin_nontemporal_store(rowsum[e], &out[g + e]);
    }
}

extern "C" void kernel_launch(void* const* d_in, const int* in_sizes, int n_in,
                              void* d_out, int out_size, void* d_ws, size_t ws_size,
                              hipStream_t stream) {
    (void)n_in; (void)ws_size; (void)out_size;
    const float* x      = (const float*)d_in[0];
    const float* conv_w = (const float*)d_in[1];
    const float* lin_w  = (const float*)d_in[2];
    float* out = (float*)d_out;

    _Float16* wsB = (_Float16*)d_ws;
    float*    wsL = (float*)((char*)d_ws + WS_LIN_OFF);

    int rows   = in_sizes[0] / K_DIM;          // 262144
    int blocks = rows / ROWS_PER_BLOCK;        // 2048
    size_t shmem = (size_t)(LB_HALFS + LA_HALFS) * sizeof(_Float16); // 67584 B

    bnn_prep_kernel<<<1, THREADS, 0, stream>>>(conv_w, lin_w, wsB, wsL);
    bnn_fused_kernel<<<blocks, THREADS, shmem, stream>>>(x, wsB, wsL, out);
}
